// Qwen3Attention_67310727463218
// MI455X (gfx1250) — compile-verified, matrix operand order checked by reading, not run on hardware
//
#include <hip/hip_runtime.h>

#define B_  2
#define S_  2048
#define D_  2048
#define H_  16
#define KV_ 4
#define G_  4
#define DK_ 128

typedef __attribute__((ext_vector_type(16))) __bf16 v16bf;
typedef __attribute__((ext_vector_type(8)))  __bf16 v8bf;
typedef __attribute__((ext_vector_type(8)))  float  v8f;
typedef int v4i_vs __attribute__((vector_size(16)));   // matches builtin param type

typedef __attribute__((address_space(1))) v4i_vs* gv4i_p;
typedef __attribute__((address_space(3))) v4i_vs* lv4i_p;

__device__ inline __bf16 f2bf(float f) {
    union { float f; unsigned u; } v; v.f = f;
    unsigned r = (v.u + 0x7FFFu + ((v.u >> 16) & 1u)) >> 16;
    union { unsigned short s; __bf16 b; } o; o.s = (unsigned short)r;
    return o.b;
}

__device__ inline v16bf cat8(v8bf lo, v8bf hi) {
    return __builtin_shufflevector(lo, hi, 0,1,2,3,4,5,6,7,8,9,10,11,12,13,14,15);
}

__device__ inline v8f wmma_bf16(v16bf a, v16bf b, v8f c) {
    return __builtin_amdgcn_wmma_f32_16x16x32_bf16(false, a, false, b, (short)0, c, false, false);
}

// CDNA5 async global->LDS copy (ASYNCcnt-tracked), with portable fallback.
__device__ inline void async_copy16(const __bf16* g, __bf16* l) {
#if __has_builtin(__builtin_amdgcn_global_load_async_to_lds_b128)
    __builtin_amdgcn_global_load_async_to_lds_b128(
        (gv4i_p)(g), (lv4i_p)(l), 0, 0);
#else
    *(v8bf*)l = *(const v8bf*)g;
#endif
}

__device__ inline void wait_async0() {
#if __has_builtin(__builtin_amdgcn_s_wait_asynccnt)
    __builtin_amdgcn_s_wait_asynccnt(0);
#elif __has_builtin(__builtin_amdgcn_global_load_async_to_lds_b128)
    asm volatile("s_wait_asynccnt 0x0" ::: "memory");
#endif
}

// ---------------------------------------------------------------------------
// fp32 -> bf16 conversion (contiguous)
// ---------------------------------------------------------------------------
__global__ void cvt_f32_bf16(const float* __restrict__ src, __bf16* __restrict__ dst, int n) {
    int i = blockIdx.x * blockDim.x + threadIdx.x;
    if (i < n) dst[i] = f2bf(src[i]);
}

// ---------------------------------------------------------------------------
// fp32 [K,N] -> bf16 transposed [N,K], 32x32 LDS tiles (coalesced both sides)
// ---------------------------------------------------------------------------
__global__ __launch_bounds__(256)
void cvt_transpose(const float* __restrict__ W, __bf16* __restrict__ Wt, int K, int N) {
    __shared__ float t[32][33];
    const int k0 = blockIdx.x * 32;
    const int n0 = blockIdx.y * 32;
    const int tx = threadIdx.x & 31;
    const int ty = threadIdx.x >> 5;      // 0..7
#pragma unroll
    for (int i = ty; i < 32; i += 8)
        t[i][tx] = W[(size_t)(k0 + i) * N + n0 + tx];
    __syncthreads();
#pragma unroll
    for (int i = ty; i < 32; i += 8)
        Wt[(size_t)(n0 + i) * K + k0 + tx] = f2bf(t[tx][i]);
}

// ---------------------------------------------------------------------------
// Tiled bf16 WMMA GEMM: C[M,N] = A[M,K](bf16) * Bt[N,K](bf16, pre-transposed)
// 256 threads = 8 waves; block tile 128x128, K step 32, double-buffered LDS
// filled by async global->LDS copies overlapped with WMMA compute.
// ---------------------------------------------------------------------------
__global__ __launch_bounds__(256)
void gemm_bf16(const __bf16* __restrict__ A, const __bf16* __restrict__ Bt,
               float* __restrict__ C, int M, int N, int K) {
    __shared__ __attribute__((aligned(16))) __bf16 As[2][128][32];
    __shared__ __attribute__((aligned(16))) __bf16 Bs[2][128][32]; // [n][k]

    const int tid   = threadIdx.x;
    const int wid   = tid >> 5;
    const int lane  = tid & 31;
    const int l16   = lane & 15;
    const int lhalf = lane >> 4;
    const int wm    = wid & 3;
    const int wn    = wid >> 2;
    const int m0    = blockIdx.y * 128;
    const int n0    = blockIdx.x * 128;

    v8f acc[2][4];
#pragma unroll
    for (int mt = 0; mt < 2; ++mt)
#pragma unroll
        for (int nt = 0; nt < 4; ++nt) acc[mt][nt] = {};

    auto stage = [&](int buf, int k0) {
        // A tile: 128 rows x 64B = 512 chunks of 16B; 2 per thread
#pragma unroll
        for (int c = tid; c < 512; c += 256) {
            int row = c >> 2, seg = (c & 3) * 8;
            async_copy16(A + (size_t)(m0 + row) * K + k0 + seg, &As[buf][row][seg]);
        }
#pragma unroll
        for (int c = tid; c < 512; c += 256) {
            int row = c >> 2, seg = (c & 3) * 8;
            async_copy16(Bt + (size_t)(n0 + row) * K + k0 + seg, &Bs[buf][row][seg]);
        }
    };

    stage(0, 0);
    wait_async0();
    __syncthreads();

    int buf = 0;
    for (int k0 = 0; k0 < K; k0 += 32) {
        if (k0 + 32 < K) stage(buf ^ 1, k0 + 32);

        v16bf af[2], bfb[4];
#pragma unroll
        for (int mt = 0; mt < 2; ++mt) {
            int row = wm * 32 + mt * 16 + l16;
            v8bf lo = *(const v8bf*)&As[buf][row][lhalf * 8];
            v8bf hi = *(const v8bf*)&As[buf][row][16 + lhalf * 8];
            af[mt] = cat8(lo, hi);
        }
#pragma unroll
        for (int nt = 0; nt < 4; ++nt) {
            int col = wn * 64 + nt * 16 + l16;
            v8bf lo = *(const v8bf*)&Bs[buf][col][lhalf * 8];
            v8bf hi = *(const v8bf*)&Bs[buf][col][16 + lhalf * 8];
            bfb[nt] = cat8(lo, hi);
        }
#pragma unroll
        for (int mt = 0; mt < 2; ++mt)
#pragma unroll
            for (int nt = 0; nt < 4; ++nt)
                acc[mt][nt] = wmma_bf16(af[mt], bfb[nt], acc[mt][nt]);

        wait_async0();
        __syncthreads();
        buf ^= 1;
    }

#pragma unroll
    for (int mt = 0; mt < 2; ++mt)
#pragma unroll
        for (int nt = 0; nt < 4; ++nt)
#pragma unroll
            for (int e = 0; e < 8; ++e) {
                int row = m0 + wm * 32 + mt * 16 + e + 8 * lhalf;
                int col = n0 + wn * 64 + nt * 16 + l16;
                C[(size_t)row * N + col] = acc[mt][nt][e];
            }
}

// ---------------------------------------------------------------------------
// Fused per-head RMSNorm + RoPE; emits bf16 q/k/v in [b, head, s, dk] layout.
// ---------------------------------------------------------------------------
__global__ __launch_bounds__(128)
void normrope_kernel(const float* __restrict__ qf, const float* __restrict__ kf,
                     const float* __restrict__ vf, const float* __restrict__ qw,
                     const float* __restrict__ kw,
                     __bf16* __restrict__ qb, __bf16* __restrict__ kb,
                     __bf16* __restrict__ vb) {
    const int r = blockIdx.x;        // 0 .. B*S-1
    const int b = r / S_;
    const int s = r % S_;
    const int tid = threadIdx.x;     // 0..127

    __shared__ float red[128];
    __shared__ float vals[128];

    const float invf = powf(10000.0f, -(float)(tid & 63) / 64.0f);
    const float ang  = (float)s * invf;
    const float cv   = cosf(ang);
    const float sv   = sinf(ang);

    for (int h = 0; h < H_; ++h) {
        float v = qf[(size_t)r * (H_ * DK_) + h * DK_ + tid];
        red[tid] = v * v;
        __syncthreads();
        for (int off = 64; off > 0; off >>= 1) {
            if (tid < off) red[tid] += red[tid + off];
            __syncthreads();
        }
        float rms = rsqrtf(red[0] * (1.0f / 128.0f) + 1e-6f);
        float nv  = v * rms * qw[tid];
        vals[tid] = nv;
        __syncthreads();
        float rot  = (tid < 64) ? -vals[tid + 64] : vals[tid - 64];
        float outv = nv * cv + rot * sv;
        qb[(((size_t)(b * H_ + h)) * S_ + s) * DK_ + tid] = f2bf(outv);
        __syncthreads();
    }

    for (int kv = 0; kv < KV_; ++kv) {
        float v = kf[(size_t)r * (KV_ * DK_) + kv * DK_ + tid];
        red[tid] = v * v;
        __syncthreads();
        for (int off = 64; off > 0; off >>= 1) {
            if (tid < off) red[tid] += red[tid + off];
            __syncthreads();
        }
        float rms = rsqrtf(red[0] * (1.0f / 128.0f) + 1e-6f);
        float nv  = v * rms * kw[tid];
        vals[tid] = nv;
        __syncthreads();
        float rot  = (tid < 64) ? -vals[tid + 64] : vals[tid - 64];
        float outv = nv * cv + rot * sv;
        kb[(((size_t)(b * KV_ + kv)) * S_ + s) * DK_ + tid] = f2bf(outv);
        __syncthreads();
    }

    for (int kv = 0; kv < KV_; ++kv) {
        float v = vf[(size_t)r * (KV_ * DK_) + kv * DK_ + tid];
        vb[(((size_t)(b * KV_ + kv)) * S_ + s) * DK_ + tid] = f2bf(v);
    }
}

// ---------------------------------------------------------------------------
// Flash attention (causal, GQA). 128 threads = 4 waves; each wave owns a
// 16-query tile; WG owns 64 queries and streams 32-key blocks.
// QK^T: 8 WMMAs (4 ksteps x 2 key tiles); P.V: 8 WMMAs (8 dk tiles).
// Output written directly as bf16 for the async-copy output GEMM.
// ---------------------------------------------------------------------------
__global__ __launch_bounds__(128)
void flash_attn(const __bf16* __restrict__ qb, const __bf16* __restrict__ kb,
                const __bf16* __restrict__ vb, __bf16* __restrict__ attn_out) {
    const int tid   = threadIdx.x;
    const int wid   = tid >> 5;
    const int lane  = tid & 31;
    const int l16   = lane & 15;
    const int lhalf = lane >> 4;

    const int bh  = blockIdx.y;          // b*H + h
    const int b   = bh / H_;
    const int h   = bh % H_;
    const int kvh = h / G_;
    const int q0  = blockIdx.x * 64;     // WG query base
    const int qw0 = q0 + wid * 16;       // wave query base

    const __bf16* qp = qb + (((size_t)(b * H_ + h)) * S_) * DK_;
    const __bf16* kp = kb + (((size_t)(b * KV_ + kvh)) * S_) * DK_;
    const __bf16* vp = vb + (((size_t)(b * KV_ + kvh)) * S_) * DK_;

    __shared__ __attribute__((aligned(16))) __bf16 Vt[128][32];   // [dk][key]
    __shared__ __attribute__((aligned(16))) __bf16 Pb[4][16][32]; // per-wave P tile

    // Q fragments (A layout): lane row = qw0 + l16
    const __bf16* qrowp = qp + (size_t)(qw0 + l16) * DK_;
    v16bf qfrag[4];
#pragma unroll
    for (int ks = 0; ks < 4; ++ks) {
        v8bf lo = *(const v8bf*)(qrowp + ks * 32 + lhalf * 8);
        v8bf hi = *(const v8bf*)(qrowp + ks * 32 + 16 + lhalf * 8);
        qfrag[ks] = cat8(lo, hi);
    }

    v8f O[8];
#pragma unroll
    for (int t = 0; t < 8; ++t) O[t] = {};
    float mi[8], li[8];
#pragma unroll
    for (int e = 0; e < 8; ++e) { mi[e] = -1e30f; li[e] = 0.0f; }

    const float scale = 0.08838834764831845f; // 1/sqrt(128)
    const int nkb = (q0 >> 5) + 2;            // key blocks of 32 covering q0+63

    for (int kbIdx = 0; kbIdx < nkb; ++kbIdx) {
        const int kbase = kbIdx * 32;

        // stage V block transposed: Vt[dk][key]
        __syncthreads();
#pragma unroll
        for (int i = tid; i < 32 * 128; i += 128) {
            int key = i >> 7, dk = i & 127;
            Vt[dk][key] = vp[(size_t)(kbase + key) * DK_ + dk];
        }
        __syncthreads();

        // scores: two 16-key tiles
        v8f s0 = {}, s1 = {};
        const __bf16* k0p = kp + (size_t)(kbase + l16) * DK_;
        const __bf16* k1p = kp + (size_t)(kbase + 16 + l16) * DK_;
#pragma unroll
        for (int ks = 0; ks < 4; ++ks) {
            v8bf lo0 = *(const v8bf*)(k0p + ks * 32 + lhalf * 8);
            v8bf hi0 = *(const v8bf*)(k0p + ks * 32 + 16 + lhalf * 8);
            s0 = wmma_bf16(qfrag[ks], cat8(lo0, hi0), s0);
            v8bf lo1 = *(const v8bf*)(k1p + ks * 32 + lhalf * 8);
            v8bf hi1 = *(const v8bf*)(k1p + ks * 32 + 16 + lhalf * 8);
            s1 = wmma_bf16(qfrag[ks], cat8(lo1, hi1), s1);
        }

        float alpha[8];
#pragma unroll
        for (int e = 0; e < 8; ++e) {
            const int m  = e + 8 * lhalf;
            const int qg = qw0 + m;
            float x0 = s0[e] * scale;
            float x1 = s1[e] * scale;
            if (kbase + l16 > qg)       x0 = -1e30f;
            if (kbase + 16 + l16 > qg)  x1 = -1e30f;
            float rm = fmaxf(x0, x1);
#pragma unroll
            for (int msk = 8; msk >= 1; msk >>= 1)
                rm = fmaxf(rm, __shfl_xor(rm, msk, 32));
            float mnew = fmaxf(mi[e], rm);
            alpha[e] = __expf(mi[e] - mnew);
            float p0 = __expf(x0 - mnew);
            float p1 = __expf(x1 - mnew);
            s0[e] = p0; s1[e] = p1;
            float rs = p0 + p1;
#pragma unroll
            for (int msk = 8; msk >= 1; msk >>= 1)
                rs += __shfl_xor(rs, msk, 32);
            li[e] = li[e] * alpha[e] + rs;
            mi[e] = mnew;
        }

        // rescale accumulators
#pragma unroll
        for (int t = 0; t < 8; ++t)
#pragma unroll
            for (int e = 0; e < 8; ++e) O[t][e] *= alpha[e];

        // C-layout P -> LDS -> A-fragment layout (intra-wave LDS is in-order)
#pragma unroll
        for (int e = 0; e < 8; ++e) {
            int m = e + 8 * lhalf;
            Pb[wid][m][l16]      = f2bf(s0[e]);
            Pb[wid][m][16 + l16] = f2bf(s1[e]);
        }
        v8bf plo = *(const v8bf*)&Pb[wid][l16][lhalf * 8];
        v8bf phi = *(const v8bf*)&Pb[wid][l16][16 + lhalf * 8];
        v16bf pfrag = cat8(plo, phi);

        // P.V over 8 dk tiles
#pragma unroll
        for (int t = 0; t < 8; ++t) {
            v8bf lo = *(const v8bf*)&Vt[t * 16 + l16][lhalf * 8];
            v8bf hi = *(const v8bf*)&Vt[t * 16 + l16][16 + lhalf * 8];
            O[t] = wmma_bf16(pfrag, cat8(lo, hi), O[t]);
        }
    }

    // epilogue: normalize and store bf16 [b*S + q, h*128 + dk]
#pragma unroll
    for (int t = 0; t < 8; ++t)
#pragma unroll
        for (int e = 0; e < 8; ++e) {
            int m  = e + 8 * lhalf;
            int qg = qw0 + m;
            float ov = O[t][e] / li[e];
            attn_out[((size_t)(b * S_ + qg)) * (H_ * DK_) + h * DK_ + t * 16 + l16] = f2bf(ov);
        }
}

// ---------------------------------------------------------------------------
extern "C" void kernel_launch(void* const* d_in, const int* in_sizes, int n_in,
                              void* d_out, int out_size, void* d_ws, size_t ws_size,
                              hipStream_t stream) {
    (void)in_sizes; (void)n_in; (void)out_size; (void)ws_size;

    const float* x  = (const float*)d_in[0];
    const float* Wq = (const float*)d_in[1];
    const float* Wk = (const float*)d_in[2];
    const float* Wv = (const float*)d_in[3];
    const float* Wo = (const float*)d_in[4];
    const float* qw = (const float*)d_in[5];
    const float* kw = (const float*)d_in[6];
    float* out = (float*)d_out;

    const int M   = B_ * S_;        // 4096
    const int Nq  = H_ * DK_;       // 2048
    const int Nkv = KV_ * DK_;      // 512

    char* wsb = (char*)d_ws;
    size_t off = 0;
    auto take = [&](size_t bytes) -> char* {
        char* p = wsb + off;
        off += (bytes + 255) & ~(size_t)255;
        return p;
    };

    __bf16* x_bf   = (__bf16*)take((size_t)M * D_   * 2);
    __bf16* Wqt    = (__bf16*)take((size_t)Nq * D_  * 2);   // [Nq][D]
    __bf16* Wkt    = (__bf16*)take((size_t)Nkv * D_ * 2);   // [Nkv][D]
    __bf16* Wvt    = (__bf16*)take((size_t)Nkv * D_ * 2);   // [Nkv][D]
    __bf16* Wot    = (__bf16*)take((size_t)D_ * Nq  * 2);   // [D][Nq]
    float*  qf     = (float*) take((size_t)M * Nq  * 4);
    float*  kf     = (float*) take((size_t)M * Nkv * 4);
    float*  vf     = (float*) take((size_t)M * Nkv * 4);
    __bf16* qbf    = (__bf16*)take((size_t)M * Nq  * 2);
    __bf16* kbf    = (__bf16*)take((size_t)M * Nkv * 2);
    __bf16* vbf    = (__bf16*)take((size_t)M * Nkv * 2);
    __bf16* attnbf = (__bf16*)take((size_t)M * Nq  * 2);

    // 1) activation conversion + weight transpose-conversion
    {
        int nx = M * D_;
        cvt_f32_bf16<<<(nx + 255) / 256, 256, 0, stream>>>(x, x_bf, nx);
        cvt_transpose<<<dim3(D_ / 32, Nq  / 32), 256, 0, stream>>>(Wq, Wqt, D_, Nq);
        cvt_transpose<<<dim3(D_ / 32, Nkv / 32), 256, 0, stream>>>(Wk, Wkt, D_, Nkv);
        cvt_transpose<<<dim3(D_ / 32, Nkv / 32), 256, 0, stream>>>(Wv, Wvt, D_, Nkv);
        cvt_transpose<<<dim3(Nq / 32, D_  / 32), 256, 0, stream>>>(Wo, Wot, Nq, D_);
    }

    // 2) QKV projections (WMMA GEMMs, async-LDS double buffered)
    gemm_bf16<<<dim3(Nq  / 128, M / 128), 256, 0, stream>>>(x_bf, Wqt, qf, M, Nq,  D_);
    gemm_bf16<<<dim3(Nkv / 128, M / 128), 256, 0, stream>>>(x_bf, Wkt, kf, M, Nkv, D_);
    gemm_bf16<<<dim3(Nkv / 128, M / 128), 256, 0, stream>>>(x_bf, Wvt, vf, M, Nkv, D_);

    // 3) per-head RMSNorm + RoPE, repack to bf16 [b, head, s, dk]
    normrope_kernel<<<M, 128, 0, stream>>>(qf, kf, vf, qw, kw, qbf, kbf, vbf);

    // 4) causal GQA flash attention (WMMA)
    flash_attn<<<dim3(S_ / 64, B_ * H_), 128, 0, stream>>>(qbf, kbf, vbf, attnbf);

    // 5) output projection (WMMA GEMM) -> d_out fp32
    gemm_bf16<<<dim3(D_ / 128, M / 128), 256, 0, stream>>>(attnbf, Wot, out, M, D_, Nq);
}
